// DGCNN_532575944998
// MI455X (gfx1250) — matrix operand order repaired and checked by reading, not compiled, not run on hardware
//
#include <hip/hip_runtime.h>
#include <math.h>

typedef __attribute__((ext_vector_type(16))) __bf16 v16bf;
typedef __attribute__((ext_vector_type(8)))  __bf16 v8bf;
typedef __attribute__((ext_vector_type(8)))  float  v8f;

#define NPTS   16384
#define PPC    2048
#define NCLOUD 8
#define KNN    20
#define NEDGE  (NPTS * KNN)
#define BN_EPS 1e-5f

#define BM 256      // rows per block (8 waves x 32 rows)
#define BNW 64      // cols per block
#define KT 32       // K tile

enum { A_F32 = 0, A_BF16BN = 1, A_EDGE = 2 };

union FragB { v16bf v; v8bf h[2]; };
union Pack8 { v8bf v; __bf16 h[8]; };

// ---- CDNA5 async global->LDS copy (ASYNCcnt-tracked), per ISA 08_async_tensor ----
__device__ __forceinline__ void async_ld_b128(unsigned int lds_off, const void* gaddr)
{
    asm volatile("global_load_async_to_lds_b128 %0, %1, off"
                 :: "v"(lds_off), "v"(gaddr) : "memory");
}
__device__ __forceinline__ void wait_async0()
{
    asm volatile("s_wait_asynccnt 0x0" ::: "memory");
}

// ---------------------------------------------------------------------------
// bf16 WMMA GEMM: C[Nrows x Nout] = act(A)[Nrows x Ka] * W[Ka x Nout] + bias
//  - Wt: bf16 weights pre-transposed & zero-padded to [Npad][Kpad] (Kpad=ceil32)
//  - A modes: fp32 / bf16 with fused BN+ReLU / fused edge gather [x_i | x_j-x_i]
//  - STATS: store bf16 C and accumulate per-column sum/sumsq for BatchNorm
//  - per wave: 32x64 tile = 2 A-frags x 4 B-frags = 8 WMMAs per K step
// ---------------------------------------------------------------------------
template<int AMODE, bool STATS>
__global__ __launch_bounds__(256)
void wmma_gemm_kernel(const float* __restrict__ Af,
                      const __bf16* __restrict__ Ab,
                      const float* __restrict__ bnscale,
                      const float* __restrict__ bnshift,
                      const int* __restrict__ eidx,
                      int dfeat, int lda,
                      const __bf16* __restrict__ Wt,   // [Npad][Kpad], transposed
                      int Ka, int Nout,
                      const float* __restrict__ bias,
                      __bf16* __restrict__ Cb,
                      float* __restrict__ Cf,
                      int ldc,
                      float* __restrict__ gstats,
                      int Nrows)
{
    __shared__ __align__(16) __bf16 As[BM * KT];   // 16 KB
    __shared__ __align__(16) __bf16 Bs[BNW * KT];  //  4 KB, layout [n][k]
    __shared__ float csum[BNW];
    __shared__ float csq[BNW];

    const int tid  = threadIdx.x;
    const int row0 = blockIdx.x * BM;
    const int n0   = blockIdx.y * BNW;
    const int kpad = (Ka + 31) & ~31;

    if (STATS && tid < BNW) { csum[tid] = 0.f; csq[tid] = 0.f; }

    v8f acc[2][4] = {};

    const int lane = tid & 31;
    const int wv   = tid >> 5;
    const int mr   = lane & 15;
    const int kh   = lane >> 4;

    // per-thread async B copy: one b128 (8 bf16 along K) per K step
    const int bn = tid >> 2;           // 0..63 : column within tile
    const int bc = (tid & 3) * 8;      // 0,8,16,24 : K chunk
    const unsigned int bsBase = (unsigned int)(size_t)(&Bs[0]);
    const unsigned int bsOff  = bsBase + (unsigned int)(bn * KT + bc) * 2u;
    const __bf16* wrow = Wt + (size_t)(n0 + bn) * kpad + bc;

    const int nk = (Ka + KT - 1) / KT;
    for (int kt = 0; kt < nk; ++kt) {
        const int kk = kt * KT;
        // ---- async-stage B tile (weights, contiguous global -> contiguous LDS) ----
        async_ld_b128(bsOff, (const void*)(wrow + kk));
        // ---- stage A tile (BM x KT) as packed v8bf chunks ----
        #pragma unroll
        for (int t = 0; t < 4; ++t) {
            const int c8 = tid + t * 256;        // chunk id, 4 chunks per row
            const int r  = c8 >> 2;
            const int kb = (c8 & 3) * 8;
            const int gr = row0 + r;
            Pack8 pk;
            if (AMODE == A_EDGE) {
                const int p = gr / KNN;
                const int j = eidx[gr];
                #pragma unroll
                for (int u = 0; u < 8; ++u) {
                    const int gk = kk + kb + u;
                    float v = 0.f;
                    if (gk < Ka && gr < Nrows) {
                        if (gk < dfeat) v = Af[(size_t)p * lda + gk];
                        else {
                            const int c = gk - dfeat;
                            v = Af[(size_t)j * lda + c] - Af[(size_t)p * lda + c];
                        }
                    }
                    pk.h[u] = (__bf16)v;
                }
            } else {
                #pragma unroll
                for (int u = 0; u < 8; ++u) {
                    const int gk = kk + kb + u;
                    float v = 0.f;
                    if (gk < Ka && gr < Nrows) {
                        if (AMODE == A_F32) {
                            v = Af[(size_t)gr * lda + gk];
                        } else {
                            v = (float)Ab[(size_t)gr * lda + gk];
                            v = fmaf(v, bnscale[gk], bnshift[gk]);
                            v = v > 0.f ? v : 0.f;
                        }
                    }
                    pk.h[u] = (__bf16)v;
                }
            }
            *reinterpret_cast<v8bf*>(&As[r * KT + kb]) = pk.v;
        }
        if ((AMODE != A_EDGE) && (kt + 1 < nk)) {
            // warm L2 for the next K tile of A (global_prefetch_b8)
            const char* pa = (AMODE == A_F32)
                ? (const char*)(Af + (size_t)(row0 + (tid >> 1)) * lda + kk + KT)
                : (const char*)(Ab + (size_t)(row0 + (tid >> 1)) * lda + kk + KT);
            __builtin_prefetch(pa, 0, 1);
        }
        wait_async0();
        __syncthreads();

        // ---- fragments per wave32 ISA layout; 8 WMMAs per K step ----
        FragB a0, a1;
        const __bf16* ar0 = As + (wv * 32 + mr) * KT;
        const __bf16* ar1 = As + (wv * 32 + 16 + mr) * KT;
        a0.h[0] = *reinterpret_cast<const v8bf*>(ar0 + kh * 8);
        a0.h[1] = *reinterpret_cast<const v8bf*>(ar0 + 16 + kh * 8);
        a1.h[0] = *reinterpret_cast<const v8bf*>(ar1 + kh * 8);
        a1.h[1] = *reinterpret_cast<const v8bf*>(ar1 + 16 + kh * 8);
        #pragma unroll
        for (int f = 0; f < 4; ++f) {
            FragB b;
            const __bf16* brow = Bs + (f * 16 + mr) * KT + kh * 8;
            b.h[0] = *reinterpret_cast<const v8bf*>(brow);
            b.h[1] = *reinterpret_cast<const v8bf*>(brow + 16);
            acc[0][f] = __builtin_amdgcn_wmma_f32_16x16x32_bf16(
                false, a0.v, false, b.v, (short)0, acc[0][f], false, false);
            acc[1][f] = __builtin_amdgcn_wmma_f32_16x16x32_bf16(
                false, a1.v, false, b.v, (short)0, acc[1][f], false, false);
        }
        __syncthreads();
    }

    // ---- epilogue: bias (+ optional bf16 store + column stats) ----
    #pragma unroll
    for (int f = 0; f < 4; ++f) {
        const int gn = n0 + f * 16 + mr;
        const float bs = (gn < Nout) ? bias[gn] : 0.f;
        float lsum = 0.f, lsq = 0.f;
        #pragma unroll
        for (int half = 0; half < 2; ++half) {
            #pragma unroll
            for (int r = 0; r < 8; ++r) {
                const int gm = row0 + wv * 32 + half * 16 + kh * 8 + r;
                const float v = acc[half][f][r] + bs;
                if (gm < Nrows && gn < Nout) {
                    if (STATS) {
                        Cb[(size_t)gm * ldc + gn] = (__bf16)v;
                        lsum += v;
                        lsq  += v * v;
                    } else {
                        Cf[(size_t)gm * ldc + gn] = v;
                    }
                }
            }
        }
        if (STATS) {
            atomicAdd(&csum[f * 16 + mr], lsum);
            atomicAdd(&csq[f * 16 + mr], lsq);
        }
    }
    if (STATS) {
        __syncthreads();
        if (tid < BNW) {
            const int gn = n0 + tid;
            if (gn < Nout) {
                atomicAdd(&gstats[gn], csum[tid]);
                atomicAdd(&gstats[512 + gn], csq[tid]);
            }
        }
    }
}

// ---------------------------------------------------------------------------
// KNN: thread owns one query (features in registers, D compile-time),
// candidates tiled through LDS, top-20 via unrolled replace-worst (self incl.)
// ---------------------------------------------------------------------------
template<int D>
__global__ __launch_bounds__(64)
void knn_kernel(const float* __restrict__ X, int ldx, int* __restrict__ Idx)
{
    __shared__ float Cand[64 * D];
    const int b    = blockIdx.y;
    const int base = b * PPC;
    const int q    = blockIdx.x * 64 + threadIdx.x;

    float qf[D];
    #pragma unroll
    for (int c = 0; c < D; ++c) qf[c] = X[(size_t)(base + q) * ldx + c];

    float bd[KNN];
    int   bi[KNN];
    #pragma unroll
    for (int t = 0; t < KNN; ++t) { bd[t] = 3.4e38f; bi[t] = base + q; }
    float worst = 3.4e38f;
    int   wslot = 0;

    for (int j0 = 0; j0 < PPC; j0 += 64) {
        __syncthreads();
        for (int e = threadIdx.x; e < 64 * D; e += 64) {
            const int r = e / D;
            const int c = e - r * D;
            Cand[e] = X[(size_t)(base + j0 + r) * ldx + c];
        }
        __syncthreads();
        for (int jj = 0; jj < 64; ++jj) {
            const float* cp = &Cand[jj * D];
            float d2 = 0.f;
            #pragma unroll
            for (int c = 0; c < D; ++c) {
                const float df = qf[c] - cp[c];
                d2 = fmaf(df, df, d2);
            }
            if (d2 < worst) {
                #pragma unroll
                for (int t = 0; t < KNN; ++t)
                    if (t == wslot) { bd[t] = d2; bi[t] = base + j0 + jj; }
                worst = -1.f;
                #pragma unroll
                for (int t = 0; t < KNN; ++t)
                    if (bd[t] > worst) { worst = bd[t]; wslot = t; }
            }
        }
    }
    #pragma unroll
    for (int t = 0; t < KNN; ++t) Idx[(size_t)(base + q) * KNN + t] = bi[t];
}

// ---------------------------------------------------------------------------
__global__ __launch_bounds__(256)
void attn_kernel(const float* __restrict__ X,
                 const float* __restrict__ w1, const float* __restrict__ b1,
                 const float* __restrict__ w2, const float* __restrict__ b2,
                 float* __restrict__ Hg)
{
    const int r = blockIdx.x * 256 + threadIdx.x;
    if (r >= NPTS) return;
    float x[9];
    #pragma unroll
    for (int c = 0; c < 9; ++c) x[c] = X[r * 9 + c];
    float h[32];
    #pragma unroll
    for (int o = 0; o < 32; ++o) {
        float a = b1[o];
        #pragma unroll
        for (int i = 0; i < 9; ++i) a = fmaf(x[i], w1[i * 32 + o], a);
        h[o] = a > 0.f ? a : 0.f;
    }
    #pragma unroll
    for (int o = 0; o < 9; ++o) {
        float a = b2[o];
        #pragma unroll
        for (int i = 0; i < 32; ++i) a = fmaf(h[i], w2[i * 9 + o], a);
        a = 1.f / (1.f + __expf(-a));
        Hg[r * 9 + o] = x[o] * a;
    }
}

// ---------------------------------------------------------------------------
__global__ void bnfinal_kernel(const float* __restrict__ st,
                               const float* __restrict__ g,
                               const float* __restrict__ be,
                               float* __restrict__ scsh, int cols, float invN)
{
    const int c = blockIdx.x * 256 + threadIdx.x;
    if (c >= cols) return;
    const float mean = st[c] * invN;
    const float var  = st[512 + c] * invN - mean * mean;
    const float s    = g[c] * rsqrtf(var + BN_EPS);
    scsh[c]       = s;
    scsh[512 + c] = be[c] - mean * s;
}

// ---------------------------------------------------------------------------
__global__ void maxreduce_kernel(const __bf16* __restrict__ H,
                                 const float* __restrict__ scsh,
                                 float* __restrict__ Xcat, int doCh, int colOff)
{
    const int gid = blockIdx.x * 256 + threadIdx.x;
    const int tot = NPTS * doCh;
    if (gid >= tot) return;
    const int p = gid / doCh;
    const int c = gid - p * doCh;
    const float s  = scsh[c];
    const float sh = scsh[512 + c];
    const __bf16* hp = H + (size_t)(p * KNN) * doCh + c;
    float m = 0.f;  // relu >= 0
    #pragma unroll
    for (int k = 0; k < KNN; ++k) {
        float v = fmaf((float)hp[(size_t)k * doCh], s, sh);
        if (v > m) m = v;
    }
    Xcat[(size_t)p * 448 + colOff + c] = m;
}

// transpose + zero-pad weights: W[Ka][Nout] fp32 -> Wt[Npad][Kpad] bf16
__global__ void cvt_w_kernel(const float* __restrict__ src, __bf16* __restrict__ dst,
                             int Ka, int Nout, int Kpad, int Npad)
{
    const int i = blockIdx.x * 256 + threadIdx.x;
    if (i >= Npad * Kpad) return;
    const int n = i / Kpad;
    const int k = i - n * Kpad;
    float v = (n < Nout && k < Ka) ? src[(size_t)k * Nout + n] : 0.f;
    dst[i] = (__bf16)v;
}

__global__ void zero_f32_kernel(float* __restrict__ p, int n)
{
    const int i = blockIdx.x * 256 + threadIdx.x;
    if (i < n) p[i] = 0.f;
}

// ---------------------------------------------------------------------------
extern "C" void kernel_launch(void* const* d_in, const int* in_sizes, int n_in,
                              void* d_out, int out_size, void* d_ws, size_t ws_size,
                              hipStream_t stream)
{
    (void)in_sizes; (void)n_in; (void)out_size; (void)ws_size;

    const float* x      = (const float*)d_in[0];
    const float* att_w1 = (const float*)d_in[1];
    const float* att_b1 = (const float*)d_in[2];
    const float* att_w2 = (const float*)d_in[3];
    const float* att_b2 = (const float*)d_in[4];
    const float* cw1[3] = {(const float*)d_in[5],  (const float*)d_in[13], (const float*)d_in[21]};
    const float* cb1[3] = {(const float*)d_in[6],  (const float*)d_in[14], (const float*)d_in[22]};
    const float* cg1[3] = {(const float*)d_in[7],  (const float*)d_in[15], (const float*)d_in[23]};
    const float* cbe1[3]= {(const float*)d_in[8],  (const float*)d_in[16], (const float*)d_in[24]};
    const float* cw2[3] = {(const float*)d_in[9],  (const float*)d_in[17], (const float*)d_in[25]};
    const float* cb2[3] = {(const float*)d_in[10], (const float*)d_in[18], (const float*)d_in[26]};
    const float* cg2[3] = {(const float*)d_in[11], (const float*)d_in[19], (const float*)d_in[27]};
    const float* cbe2[3]= {(const float*)d_in[12], (const float*)d_in[20], (const float*)d_in[28]};
    const float* m_w1 = (const float*)d_in[29];
    const float* m_b1 = (const float*)d_in[30];
    const float* m_g1 = (const float*)d_in[31];
    const float* m_be1= (const float*)d_in[32];
    const float* m_w2 = (const float*)d_in[33];
    const float* m_b2 = (const float*)d_in[34];
    const float* m_g2 = (const float*)d_in[35];
    const float* m_be2= (const float*)d_in[36];
    const float* m_w3 = (const float*)d_in[37];
    const float* m_b3 = (const float*)d_in[38];
    float* out = (float*)d_out;

    // ---- workspace carve-out ----
    char* ws = (char*)d_ws;
    size_t off = 0;
    auto alloc = [&](size_t bytes) -> void* {
        void* p = ws + off;
        off = (off + bytes + 255) & ~(size_t)255;
        return p;
    };
    float*  Hg    = (float*) alloc((size_t)NPTS * 9 * 4);
    float*  Xcat  = (float*) alloc((size_t)NPTS * 448 * 4);
    int*    Idx   = (int*)   alloc((size_t)NPTS * KNN * 4);
    __bf16* hA    = (__bf16*)alloc((size_t)NEDGE * 256 * 2);
    __bf16* hB    = (__bf16*)alloc((size_t)NEDGE * 256 * 2);
    float*  stats = (float*) alloc(8 * 1024 * 4);   // 8 BN slots x (sum[512], sq[512])
    float*  scsh  = (float*) alloc(8 * 1024 * 4);   // 8 BN slots x (scale[512], shift[512])

    // transposed, padded bf16 weights: [Npad][Kpad]
    const int wKa[9]   = {18, 64, 128, 128, 256, 256, 448, 512, 256};
    const int wNo[9]   = {64, 64, 128, 128, 256, 256, 512, 256, 5};
    const int wKpad[9] = {32, 64, 128, 128, 256, 256, 448, 512, 256};
    const int wNpad[9] = {64, 64, 128, 128, 256, 256, 512, 256, 64};
    __bf16* wbf[9];
    for (int i = 0; i < 9; ++i)
        wbf[i] = (__bf16*)alloc((size_t)wKpad[i] * wNpad[i] * 2);

    const int cdim[3] = {9, 64, 128};     // edgeconv input feature dims
    const int cout[3] = {64, 128, 256};   // edgeconv output dims
    const int coff[3] = {0, 64, 192};     // column offsets of x1/x2/x3 in Xcat

    // ---- zero BN stats, transpose+convert weights to bf16 ----
    zero_f32_kernel<<<(8 * 1024 + 255) / 256, 256, 0, stream>>>(stats, 8 * 1024);
    const float* wsrc[9] = {cw1[0], cw2[0], cw1[1], cw2[1], cw1[2], cw2[2], m_w1, m_w2, m_w3};
    for (int i = 0; i < 9; ++i) {
        const int n = wKpad[i] * wNpad[i];
        cvt_w_kernel<<<(n + 255) / 256, 256, 0, stream>>>(
            wsrc[i], wbf[i], wKa[i], wNo[i], wKpad[i], wNpad[i]);
    }

    // ---- attention gate ----
    attn_kernel<<<(NPTS + 255) / 256, 256, 0, stream>>>(x, att_w1, att_b1, att_w2, att_b2, Hg);

    // ---- EdgeConv layers ----
    const float invNE = 1.f / (float)NEDGE;
    for (int l = 0; l < 3; ++l) {
        const int D  = cdim[l];
        const int DO = cout[l];
        const float* Xf = (l == 0) ? Hg : (Xcat + coff[l - 1]);
        const int ldx   = (l == 0) ? 9 : 448;
        dim3 kg(PPC / 64, NCLOUD);
        if (l == 0)      knn_kernel<9  ><<<kg, 64, 0, stream>>>(Xf, ldx, Idx);
        else if (l == 1) knn_kernel<64 ><<<kg, 64, 0, stream>>>(Xf, ldx, Idx);
        else             knn_kernel<128><<<kg, 64, 0, stream>>>(Xf, ldx, Idx);

        float* st1 = stats + (size_t)(2 * l) * 1024;
        float* ss1 = scsh  + (size_t)(2 * l) * 1024;
        float* st2 = stats + (size_t)(2 * l + 1) * 1024;
        float* ss2 = scsh  + (size_t)(2 * l + 1) * 1024;
        dim3 g1(NEDGE / BM, (DO + BNW - 1) / BNW);
        wmma_gemm_kernel<A_EDGE, true><<<g1, 256, 0, stream>>>(
            Xf, nullptr, nullptr, nullptr, Idx, D, ldx,
            wbf[2 * l], 2 * D, DO, cb1[l], hA, nullptr, DO, st1, NEDGE);
        bnfinal_kernel<<<(DO + 255) / 256, 256, 0, stream>>>(st1, cg1[l], cbe1[l], ss1, DO, invNE);
        wmma_gemm_kernel<A_BF16BN, true><<<g1, 256, 0, stream>>>(
            nullptr, hA, ss1, ss1 + 512, nullptr, 0, DO,
            wbf[2 * l + 1], DO, DO, cb2[l], hB, nullptr, DO, st2, NEDGE);
        bnfinal_kernel<<<(DO + 255) / 256, 256, 0, stream>>>(st2, cg2[l], cbe2[l], ss2, DO, invNE);
        const int tot = NPTS * DO;
        maxreduce_kernel<<<(tot + 255) / 256, 256, 0, stream>>>(hB, ss2, Xcat, DO, coff[l]);
    }

    // ---- final MLP: 448 -> 512 -> 256 -> 5 ----
    const float invNP = 1.f / (float)NPTS;
    float* st6 = stats + 6 * 1024; float* ss6 = scsh + 6 * 1024;
    float* st7 = stats + 7 * 1024; float* ss7 = scsh + 7 * 1024;
    __bf16* hm1 = hA;  // reuse (16384 x 512)
    __bf16* hm2 = hB;  // reuse (16384 x 256)
    dim3 gm1(NPTS / BM, 512 / BNW);
    wmma_gemm_kernel<A_F32, true><<<gm1, 256, 0, stream>>>(
        Xcat, nullptr, nullptr, nullptr, nullptr, 0, 448,
        wbf[6], 448, 512, m_b1, hm1, nullptr, 512, st6, NPTS);
    bnfinal_kernel<<<2, 256, 0, stream>>>(st6, m_g1, m_be1, ss6, 512, invNP);
    dim3 gm2(NPTS / BM, 256 / BNW);
    wmma_gemm_kernel<A_BF16BN, true><<<gm2, 256, 0, stream>>>(
        nullptr, hm1, ss6, ss6 + 512, nullptr, 0, 512,
        wbf[7], 512, 256, m_b2, hm2, nullptr, 256, st7, NPTS);
    bnfinal_kernel<<<1, 256, 0, stream>>>(st7, m_g2, m_be2, ss7, 256, invNP);
    dim3 gm3(NPTS / BM, 1);
    wmma_gemm_kernel<A_BF16BN, false><<<gm3, 256, 0, stream>>>(
        nullptr, hm2, ss7, ss7 + 512, nullptr, 0, 256,
        wbf[8], 256, 5, m_b3, nullptr, out, 5, nullptr, NPTS);
}
